// HyperbolicMoE_87033217286171
// MI455X (gfx1250) — matrix operand order repaired and checked by reading, not compiled, not run on hardware
//
#include <hip/hip_runtime.h>
#include <hip/hip_bf16.h>
#include <math.h>

// HyperbolicMoE on gfx1250: top-2 dispatched MoE with bf16 WMMA (16x16x32),
// fp32 accumulation. Compute-bound (137 GFLOP sparse vs 550 dense); weights
// L2-resident per expert (32MB/expert vs 192MB L2). B fragments are built
// from LDS with 2x ds_load_b128 via transposed, pad-strided staging tiles.

#define T_TOK 4096
#define DIM   1024
#define NEXP  8
#define FDIM  4096
#define MAXN  0.95f
#define EPSR  1e-7f

// padded LDS strides (in ushorts); multiples of 8 keep 16B alignment for b128
#define W1T_STRIDE 520   // [32 f][512 d]   -> 16640 ushorts
#define W2T_STRIDE 40    // [512 d][32 f]   -> 20480 ushorts
#define KBUF_MAX   33280
#define WPT_STRIDE 40    // [64 n][32 k]

typedef __attribute__((ext_vector_type(16))) __bf16 v16bf;
typedef __attribute__((ext_vector_type(8)))  float  v8f;

union Frag {
  v16bf v;
  unsigned short s[16];
  uint4 q[2];
};

__device__ __forceinline__ unsigned short f2b(float f) {
  union { float f; unsigned int u; } cv; cv.f = f;
  unsigned int u = cv.u;
  u += 0x7FFFu + ((u >> 16) & 1u);          // round-to-nearest-even
  return (unsigned short)(u >> 16);
}

__device__ __forceinline__ v8f vzero8() {
  v8f z = {0.f, 0.f, 0.f, 0.f, 0.f, 0.f, 0.f, 0.f};
  return z;
}

__device__ __forceinline__ v8f wmma_bf16(const Frag& a, const Frag& b, v8f c) {
  return __builtin_amdgcn_wmma_f32_16x16x32_bf16(false, a.v, false, b.v,
                                                 (short)0, c, false, false);
}

// load a 16-contiguous-ushort B fragment (lane = column, K contiguous)
__device__ __forceinline__ void loadB(Frag& b, const unsigned short* p) {
  b.q[0] = *(const uint4*)p;
  b.q[1] = *(const uint4*)(p + 8);
}
// A fragment: 16 contiguous ushorts at k0 (contiguous K order, matches B)
__device__ __forceinline__ void loadA(Frag& a, const unsigned short* p) {
  a.q[0] = *(const uint4*)p;
  a.q[1] = *(const uint4*)(p + 8);
}

__device__ __forceinline__ float wred(float v) {
  v += __shfl_xor(v, 16, 32);
  v += __shfl_xor(v, 8, 32);
  v += __shfl_xor(v, 4, 32);
  v += __shfl_xor(v, 2, 32);
  v += __shfl_xor(v, 1, 32);
  return v;
}

// ---------------- K0: convert x,Wp to bf16; zero output ----------------
__global__ __launch_bounds__(256) void k_prep(
    const float* __restrict__ x, const float* __restrict__ Wp,
    unsigned short* __restrict__ xb, unsigned short* __restrict__ Wpb,
    float* __restrict__ out) {
  const int n4 = T_TOK * DIM / 4;
  const float4 z4 = make_float4(0.f, 0.f, 0.f, 0.f);
  for (int i = blockIdx.x * blockDim.x + threadIdx.x; i < n4;
       i += gridDim.x * blockDim.x) {
    const float4 v = ((const float4*)x)[i];
    uint2 p;
    p.x = (unsigned)f2b(v.x) | ((unsigned)f2b(v.y) << 16);
    p.y = (unsigned)f2b(v.z) | ((unsigned)f2b(v.w) << 16);
    ((uint2*)xb)[i] = p;
    ((float4*)out)[i] = z4;
    if (i < DIM * DIM / 4) {
      const float4 w = ((const float4*)Wp)[i];
      uint2 q;
      q.x = (unsigned)f2b(w.x) | ((unsigned)f2b(w.y) << 16);
      q.y = (unsigned)f2b(w.z) | ((unsigned)f2b(w.w) << 16);
      ((uint2*)Wpb)[i] = q;
    }
  }
}

// ---------------- K1: xh = x @ Wp + bp  (bf16 WMMA, fp32 out) ----------
// grid (T/64, D/64), block 256. Wave w: row tile w>>1, col tiles {2g,2g+1}.
__global__ __launch_bounds__(256) void k_proj_gemm(
    const unsigned short* __restrict__ xb, const unsigned short* __restrict__ Wpb,
    const float* __restrict__ bp, float* __restrict__ xh) {
  __shared__ __align__(16) unsigned short sB[64 * WPT_STRIDE];
  const int tid = threadIdx.x, lane = tid & 31, wv = tid >> 5;
  const int r = wv >> 1, g = wv & 1, half = lane >> 4, ln = lane & 15;
  const int row0 = blockIdx.x * 64 + 16 * r;
  const int col0 = blockIdx.y * 64;
  v8f acc0 = vzero8(), acc1 = vzero8();
  const int n0 = col0 + 32 * g + ln;
  // staging decomposition: this thread's 8 contiguous ushorts of Wp chunk
  const int sk = tid >> 3;          // k row 0..31
  const int sn = (tid & 7) * 8;     // n col, 8-wide
  for (int kk = 0; kk < 32; ++kk) {
    // stage Wp[32kk..+32, col0..+64) transposed into sB[n][k]
    {
      const uint4 w = *(const uint4*)&Wpb[(size_t)(32 * kk + sk) * DIM + col0 + sn];
      const unsigned short* w16 = (const unsigned short*)&w;
#pragma unroll
      for (int c = 0; c < 8; ++c) sB[(sn + c) * WPT_STRIDE + sk] = w16[c];
    }
    __syncthreads();
    Frag a;
    loadA(a, xb + (size_t)(row0 + ln) * DIM + 32 * kk + 16 * half);
    Frag b0, b1;
    loadB(b0, &sB[(32 * g + ln) * WPT_STRIDE + 16 * half]);
    loadB(b1, &sB[(32 * g + 16 + ln) * WPT_STRIDE + 16 * half]);
    acc0 = wmma_bf16(a, b0, acc0);
    acc1 = wmma_bf16(a, b1, acc1);
    __syncthreads();
  }
  const float bp0 = bp[n0], bp1 = bp[n0 + 16];
#pragma unroll
  for (int i = 0; i < 8; ++i) {
    const int m = row0 + i + 8 * half;
    xh[(size_t)m * DIM + n0]      = acc0[i] + bp0;
    xh[(size_t)m * DIM + n0 + 16] = acc1[i] + bp1;
  }
}

// ---------------- K2: hyperbolic routing, top-2 softmax ----------------
// grid T/8, block 256; one wave per token.
__global__ __launch_bounds__(256) void k_route(
    const float* __restrict__ xh, const float* __restrict__ cen,
    int2* __restrict__ re, float2* __restrict__ rw) {
  __shared__ float lC[NEXP * DIM];
  const int tid = threadIdx.x;
  for (int i = tid; i < NEXP * DIM; i += 256) lC[i] = cen[i];
  __syncthreads();
  const int lane = tid & 31, wv = tid >> 5;
  const int t = blockIdx.x * 8 + wv;
  float xr[32];
  float n2 = 0.f;
#pragma unroll
  for (int i = 0; i < 32; ++i) {
    xr[i] = xh[(size_t)t * DIM + i * 32 + lane];
    n2 += xr[i] * xr[i];
  }
  n2 = wred(n2);
  const float norm = fmaxf(sqrtf(n2), EPSR);
  const float sc = (norm > MAXN) ? (MAXN / norm) : 1.0f;
  const float xn2 = fminf(n2 * sc * sc, 0.99f);
  float v0 = -1e30f, v1 = -1e30f;
  int i0 = 0, i1 = 0;
  for (int e = 0; e < NEXP; ++e) {
    float d = 0.f, c2 = 0.f;
#pragma unroll
    for (int i = 0; i < 32; ++i) {
      const float c = lC[e * DIM + i * 32 + lane];
      d += xr[i] * c;
      c2 += c * c;
    }
    d = wred(d);
    c2 = wred(c2);
    const float cn = fmaxf(sqrtf(c2), EPSR);
    const float cs = (cn > MAXN) ? (MAXN / cn) : 1.0f;
    const float cn2 = fminf(c2 * cs * cs, 0.99f);
    const float dot = d * sc * cs;
    const float diff = fmaxf(xn2 + cn2 - 2.0f * dot, 0.f);
    const float den = (1.f - xn2) * (1.f - cn2) + EPSR;
    const float arg = fmaxf(1.f + 2.f * diff / den, 1.f + EPSR);
    const float dist = logf(arg + sqrtf(arg * arg - 1.f));
    const float lg = -dist;
    if (lg > v0) { v1 = v0; i1 = i0; v0 = lg; i0 = e; }
    else if (lg > v1) { v1 = lg; i1 = e; }
  }
  if (lane == 0) {
    const float ew = expf(v1 - v0);
    const float w0 = 1.f / (1.f + ew);
    re[t] = make_int2(i0, i1);
    rw[t] = make_float2(w0, 1.f - w0);
  }
}

// ---------------- K3: build per-expert token lists (deterministic) -----
__global__ __launch_bounds__(256) void k_dispatch(
    const int2* __restrict__ re, const float2* __restrict__ rw,
    int* __restrict__ tlist, float* __restrict__ twt,
    int* __restrict__ gcnt, int* __restrict__ goff) {
  __shared__ int lcnt[NEXP], loff[NEXP];
  const int tid = threadIdx.x, lane = tid & 31, e = tid >> 5;
  const unsigned lmask = (1u << lane) - 1u;
  int cnt = 0;
  for (int base = 0; base < T_TOK; base += 32) {
    const int t = base + lane;
    const int2 et = re[t];
    const bool m = (et.x == e) || (et.y == e);
    cnt += __popc((unsigned)__ballot(m));
  }
  if (lane == 0) lcnt[e] = cnt;
  __syncthreads();
  if (tid == 0) {
    int o = 0;
    for (int i = 0; i < NEXP; ++i) { loff[i] = o; o += lcnt[i]; }
  }
  __syncthreads();
  int off = loff[e];
  for (int base = 0; base < T_TOK; base += 32) {
    const int t = base + lane;
    const int2 et = re[t];
    const float2 wt = rw[t];
    const bool m0 = (et.x == e);
    const bool m = m0 || (et.y == e);
    const unsigned b = (unsigned)__ballot(m);
    if (m) {
      const int p = off + __popc(b & lmask);
      tlist[p] = t;
      twt[p] = m0 ? wt.x : wt.y;
    }
    off += __popc(b);
  }
  if (lane == 0) { gcnt[e] = lcnt[e]; goff[e] = loff[e]; }
}

// ---------------- K4: fused expert FFN on gathered tokens --------------
// grid (64 tiles, 8 experts), block 256 (8 waves). 64 tokens/block,
// F chunked by 32; one 65KB... (40KB) LDS buffer time-shared between the
// transposed W1 / W2 staging tiles. Wave w: row tile r=w>>1, col group
// g=w&1 (512 d-cols = 32 fp32 WMMA accumulator tiles, fully unrolled).
__global__ __launch_bounds__(256) void k_expert(
    const unsigned short* __restrict__ xb,
    const float* __restrict__ W1, const float* __restrict__ b1,
    const float* __restrict__ W2, const float* __restrict__ b2,
    const int* __restrict__ tlist, const float* __restrict__ twt,
    const int* __restrict__ gcnt, const int* __restrict__ goff,
    float* __restrict__ out) {
  const int e = blockIdx.y;
  const int cnt = gcnt[e];
  const int base = blockIdx.x * 64;
  if (base >= cnt) return;
  const int off = goff[e];

  __shared__ __align__(16) unsigned short buf[KBUF_MAX]; // W1T 33KB / W2T 40KB... union (65KB alloc)
  __shared__ __align__(16) unsigned short lH[64 * 32];   // 4KB h tile (bf16)
  __shared__ int   lTok[64];
  __shared__ float lWt[64];

  const int tid = threadIdx.x;
  if (tid < 64) {
    const int idx = base + tid;
    lTok[tid] = (idx < cnt) ? tlist[off + idx] : -1;
    lWt[tid]  = (idx < cnt) ? twt[off + idx] : 0.f;
  }
  __syncthreads();

  const int lane = tid & 31, wv = tid >> 5;
  const int r = wv >> 1, g = wv & 1, half = lane >> 4, ln = lane & 15;

  v8f acc[32];
#pragma unroll
  for (int i = 0; i < 32; ++i) acc[i] = vzero8();

  const float* W1e = W1 + (size_t)e * DIM * FDIM;
  const float* W2e = W2 + (size_t)e * FDIM * DIM;
  const int mtok = lTok[16 * r + ln];
  const size_t xrow = (size_t)(mtok < 0 ? 0 : mtok) * DIM;

  for (int f0 = 0; f0 < FDIM; f0 += 32) {
    v8f hacc = vzero8();
#pragma unroll
    for (int kh = 0; kh < 2; ++kh) {
      // stage W1[512*kh..+512, f0..+32) transposed bf16: buf[f*520 + d]
      for (int i4 = tid; i4 < 4096; i4 += 256) {
        const int flat = i4 * 4;
        const int d = flat >> 5;        // 0..511
        const int f = flat & 31;        // multiple of 4
        const float4 w = *(const float4*)&W1e[(size_t)(512 * kh + d) * FDIM + f0 + f];
        buf[(f + 0) * W1T_STRIDE + d] = f2b(w.x);
        buf[(f + 1) * W1T_STRIDE + d] = f2b(w.y);
        buf[(f + 2) * W1T_STRIDE + d] = f2b(w.z);
        buf[(f + 3) * W1T_STRIDE + d] = f2b(w.w);
      }
      // hint: next chunk's W1 rows (gfx1250 global_prefetch)
      if (f0 + 32 < FDIM)
        __builtin_prefetch(&W1e[(size_t)(512 * kh + tid * 2) * FDIM + f0 + 32], 0, 1);
      __syncthreads();
#pragma unroll 4
      for (int kk = 0; kk < 16; ++kk) {
        Frag a;
        loadA(a, xb + xrow + 512 * kh + 32 * kk + 16 * half);
        Frag b;
        loadB(b, &buf[(16 * g + ln) * W1T_STRIDE + 32 * kk + 16 * half]);
        hacc = wmma_bf16(a, b, hacc);
      }
      __syncthreads();
    }
    // h epilogue: bias + exact GELU, bf16 into lH
    const float bv = b1[e * FDIM + f0 + 16 * g + ln];
#pragma unroll
    for (int i = 0; i < 8; ++i) {
      float h = hacc[i] + bv;
      h = 0.5f * h * (1.0f + erff(h * 0.70710678118654752f));
      lH[(16 * r + i + 8 * half) * 32 + 16 * g + ln] = f2b(h);
    }
    __syncthreads();
    // GEMM2 A fragment from lH (K=32 chunk)
    Frag a2;
    loadA(a2, &lH[(16 * r + ln) * 32 + 16 * half]);
#pragma unroll
    for (int dh = 0; dh < 2; ++dh) {
      // stage W2[f0..+32, 512*dh..+512) transposed bf16: buf[d*40 + f]
      for (int i4 = tid; i4 < 4096; i4 += 256) {
        const int flat = i4 * 4;
        const int f = flat >> 9;        // 0..31
        const int d = flat & 511;       // multiple of 4
        const float4 w = *(const float4*)&W2e[(size_t)(f0 + f) * DIM + 512 * dh + d];
        buf[(d + 0) * W2T_STRIDE + f] = f2b(w.x);
        buf[(d + 1) * W2T_STRIDE + f] = f2b(w.y);
        buf[(d + 2) * W2T_STRIDE + f] = f2b(w.z);
        buf[(d + 3) * W2T_STRIDE + f] = f2b(w.w);
      }
      if (f0 + 32 < FDIM)
        __builtin_prefetch(&W2e[(size_t)(f0 + 32 + (tid >> 3)) * DIM + 512 * dh + (tid & 7) * 64], 0, 1);
      __syncthreads();
#pragma unroll
      for (int jj = 0; jj < 16; ++jj) {
        Frag b;
        loadB(b, &buf[(256 * g + 16 * jj + ln) * W2T_STRIDE + 16 * half]);
        acc[dh * 16 + jj] = wmma_bf16(a2, b, acc[dh * 16 + jj]);
      }
      __syncthreads();
    }
  }
  // out += w * (acc + b2): exactly two commutative fp32 adds per element
#pragma unroll
  for (int t2 = 0; t2 < 32; ++t2) {
    const int dh = t2 >> 4, jj = t2 & 15;
    const int n = 512 * dh + 256 * g + 16 * jj + ln;
    const float b2v = b2[e * DIM + n];
#pragma unroll
    for (int i = 0; i < 8; ++i) {
      const int mrow = 16 * r + i + 8 * half;
      const int tok = lTok[mrow];
      if (tok >= 0) {
        atomicAdd(&out[(size_t)tok * DIM + n], lWt[mrow] * (acc[t2][i] + b2v));
      }
    }
  }
}

// ---------------------------------------------------------------------
extern "C" void kernel_launch(void* const* d_in, const int* in_sizes, int n_in,
                              void* d_out, int out_size, void* d_ws, size_t ws_size,
                              hipStream_t stream) {
  (void)in_sizes; (void)n_in; (void)out_size; (void)ws_size;
  const float* x   = (const float*)d_in[0];
  const float* cen = (const float*)d_in[1];
  const float* Wp  = (const float*)d_in[2];
  const float* bp  = (const float*)d_in[3];
  const float* W1  = (const float*)d_in[4];
  const float* b1  = (const float*)d_in[5];
  const float* W2  = (const float*)d_in[6];
  const float* b2  = (const float*)d_in[7];
  float* out = (float*)d_out;

  char* ws = (char*)d_ws;
  unsigned short* xb  = (unsigned short*)(ws + 0);            // 8 MB
  unsigned short* Wpb = (unsigned short*)(ws + 8388608);      // 2 MB
  float*          xh  = (float*)(ws + 10485760);              // 16 MB
  int2*           re  = (int2*)(ws + 27262976);               // 32 KB
  float2*         rw  = (float2*)(ws + 27295744);             // 32 KB
  int*            tl  = (int*)(ws + 27328512);                // 32 KB
  float*          tw  = (float*)(ws + 27361280);              // 32 KB
  int*            gcnt = (int*)(ws + 27394048);
  int*            goff = (int*)(ws + 27394080);

  k_prep<<<1024, 256, 0, stream>>>(x, Wp, xb, Wpb, out);
  k_proj_gemm<<<dim3(T_TOK / 64, DIM / 64), 256, 0, stream>>>(xb, Wpb, bp, xh);
  k_route<<<T_TOK / 8, 256, 0, stream>>>(xh, cen, re, rw);
  k_dispatch<<<1, 256, 0, stream>>>(re, rw, tl, tw, gcnt, goff);
  k_expert<<<dim3(64, NEXP), 256, 0, stream>>>(xb, W1, b1, W2, b2,
                                               tl, tw, gcnt, goff, out);
}
